// SphereLoss_44968307589188
// MI455X (gfx1250) — matrix operand order repaired and checked by reading, not compiled, not run on hardware
//
#include <hip/hip_runtime.h>
#include <hip/hip_bf16.h>
#include <math.h>

// ---------------------------------------------------------------------------
// SphereFace-style loss, MI455X (gfx1250, wave32, WMMA).
//   x  [8192 x 128]  f32
//   W  [128  x 10000] f32
//   out[8192 x 10000] f32 = S_m / rowsum(S_m),
//   S_m = Xn*(2*(S/(Xn+eps))^2 - 1),  S = x @ (W / colnorm(W))
//
// Strategy: f16 WMMA GEMM (f32 accum). rowsum(S_m) = Xn*(2*q/(Xn+eps)^2 - C)
// with q = sum_n S_n^2, so pass 1 accumulates q per row with f32 atomics and
// pass 2 fuses the epilogue into the single 328 MB output write.
// ---------------------------------------------------------------------------

typedef __attribute__((ext_vector_type(16))) _Float16 v16h;
typedef __attribute__((ext_vector_type(8)))  _Float16 v8h;
typedef __attribute__((ext_vector_type(8)))  float    v8f;

#define BATCH   8192
#define KDIM    128
#define NCLS    10000
#define NPAD    10112   /* 79*128, zero-padded columns */
#define TURB    1e-07f

// workspace layout (bytes), all 256-aligned
#define XH_OFF   0u
#define XH_BYTES (BATCH * KDIM * 2u)                 // 2 MB  f16 x
#define WT_OFF   (XH_OFF + XH_BYTES)
#define WT_BYTES (NPAD * KDIM * 2u)                  // 2.6 MB f16 Wn^T
#define XN_OFF   (WT_OFF + WT_BYTES)
#define XN_BYTES (BATCH * 4u)
#define QS_OFF   (XN_OFF + XN_BYTES)

__device__ __forceinline__ v16h cat8(v8h lo, v8h hi) {
    return __builtin_shufflevector(lo, hi, 0,1,2,3,4,5,6,7,8,9,10,11,12,13,14,15);
}

// A fragment (16x32 f16, M=row across lanes&15): lanes<16 hold K = k0+[0..15],
// lanes>=16 hold K = k0+[16..31].
__device__ __forceinline__ v16h load_a(const _Float16* __restrict__ xh,
                                       int row, int k0, int lane) {
    const int koff = k0 + ((lane & 16) ? 16 : 0);
    const _Float16* p = xh + row * KDIM + koff;
    v8h lo = *(const v8h*)(p);
    v8h hi = *(const v8h*)(p + 8);
    return cat8(lo, hi);
}

// B fragment (32x16 f16, N=col across lanes&15): lanes<16 hold K = k0+[0..15],
// lanes>=16 hold K = k0+[16..31]. WnT is [col][k] so this is contiguous.
__device__ __forceinline__ v16h load_b(const _Float16* __restrict__ wnt,
                                       int col, int k0, int lane) {
    const int kbase = k0 + ((lane & 16) ? 16 : 0);
    const _Float16* p = wnt + col * KDIM + kbase;
    v8h lo = *(const v8h*)(p);
    v8h hi = *(const v8h*)(p + 8);
    return cat8(lo, hi);
}

// --- prep: x -> f16 row-major, Xn = ||row||, qsum = 0.  One wave per row. ---
__global__ __launch_bounds__(256) void prep_x_kernel(
        const float* __restrict__ x, _Float16* __restrict__ xh,
        float* __restrict__ xn, float* __restrict__ qsum) {
    const int lane = threadIdx.x & 31;
    const int row  = blockIdx.x * 8 + (threadIdx.x >> 5);
    const float* xr = x + row * KDIM;
    _Float16*    xo = xh + row * KDIM;
    float ss = 0.0f;
    #pragma unroll
    for (int t = 0; t < 4; ++t) {
        float v = xr[t * 32 + lane];
        ss += v * v;
        xo[t * 32 + lane] = (_Float16)v;
    }
    #pragma unroll
    for (int off = 16; off; off >>= 1) ss += __shfl_xor(ss, off, 32);
    if (lane == 0) { xn[row] = sqrtf(ss); qsum[row] = 0.0f; }
}

// --- prep: Wn^T f16 [NPAD][128], columns normalized, pad rows zeroed. ------
__global__ __launch_bounds__(256) void prep_w_kernel(
        const float* __restrict__ W, _Float16* __restrict__ wnt) {
    const int lane = threadIdx.x & 31;
    const int col  = blockIdx.x * 8 + (threadIdx.x >> 5);   // wave per column
    _Float16* o = wnt + col * KDIM;
    if (col >= NCLS) {                                      // uniform per wave
        #pragma unroll
        for (int t = 0; t < 4; ++t) o[t * 32 + lane] = (_Float16)0.0f;
        return;
    }
    float vals[4];
    float ss = 0.0f;
    #pragma unroll
    for (int t = 0; t < 4; ++t) {
        float v = W[(t * 32 + lane) * NCLS + col];
        vals[t] = v;
        ss += v * v;
    }
    #pragma unroll
    for (int off = 16; off; off >>= 1) ss += __shfl_xor(ss, off, 32);
    const float inv = 1.0f / sqrtf(ss);
    #pragma unroll
    for (int t = 0; t < 4; ++t) o[t * 32 + lane] = (_Float16)(vals[t] * inv);
}

// --- GEMM: 128x128 tile/block, 8 waves, wave = 32(M) x 64(N). --------------
// PASS 1: qsum[row] += sum_n s^2   (f32 global atomics)
// PASS 2: out = cA*s^2 + cB  fused epilogue, single output write
template <int PASS>
__global__ __launch_bounds__(256) void gemm_kernel(
        const _Float16* __restrict__ xh, const _Float16* __restrict__ wnt,
        const float* __restrict__ xn, float* __restrict__ qsum,
        float* __restrict__ out) {
    const int lane = threadIdx.x & 31;
    const int w    = threadIdx.x >> 5;
    const int M0   = blockIdx.y * 128 + (w & 3) * 32;
    const int N0   = blockIdx.x * 128 + (w >> 2) * 64;
    const int l15  = lane & 15;

    v8f acc[2][4] = {};
    #pragma unroll
    for (int k0 = 0; k0 < KDIM; k0 += 32) {
        v16h a[2], b[4];
        #pragma unroll
        for (int mt = 0; mt < 2; ++mt)
            a[mt] = load_a(xh, M0 + mt * 16 + l15, k0, lane);
        #pragma unroll
        for (int nt = 0; nt < 4; ++nt)
            b[nt] = load_b(wnt, N0 + nt * 16 + l15, k0, lane);
        #pragma unroll
        for (int mt = 0; mt < 2; ++mt)
            #pragma unroll
            for (int nt = 0; nt < 4; ++nt)
                acc[mt][nt] = __builtin_amdgcn_wmma_f32_16x16x32_f16(
                    false, a[mt], false, b[nt], (short)0, acc[mt][nt],
                    false, false);
    }

    // C layout: VGPR r -> (M = r + (lane>=16 ? 8 : 0), N = lane&15)
    if (PASS == 1) {
        #pragma unroll
        for (int mt = 0; mt < 2; ++mt) {
            #pragma unroll
            for (int r = 0; r < 8; ++r) {
                float s2 = 0.0f;
                #pragma unroll
                for (int nt = 0; nt < 4; ++nt) {
                    float s = acc[mt][nt][r];
                    s2 += s * s;
                }
                #pragma unroll
                for (int off = 1; off < 16; off <<= 1)   // stays within half
                    s2 += __shfl_xor(s2, off, 32);
                if (l15 == 0) {
                    int grow = M0 + mt * 16 + r + ((lane & 16) ? 8 : 0);
                    atomicAdd(&qsum[grow], s2);
                }
            }
        }
    } else {
        #pragma unroll
        for (int mt = 0; mt < 2; ++mt) {
            #pragma unroll
            for (int r = 0; r < 8; ++r) {
                const int grow = M0 + mt * 16 + r + ((lane & 16) ? 8 : 0);
                const float X   = xn[grow];
                const float q   = qsum[grow];
                const float xe  = X + TURB;
                const float a   = 2.0f * X / (xe * xe);
                const float rs  = a * q - (float)NCLS * X;   // rowsum(S_m)
                const float inv = 1.0f / rs;
                const float cA  = a * inv;
                const float cB  = -X * inv;
                #pragma unroll
                for (int nt = 0; nt < 4; ++nt) {
                    const int col = N0 + nt * 16 + l15;
                    if (col < NCLS) {
                        float s = acc[mt][nt][r];
                        out[(long)grow * NCLS + col] = cA * s * s + cB;
                    }
                }
            }
        }
    }
}

extern "C" void kernel_launch(void* const* d_in, const int* in_sizes, int n_in,
                              void* d_out, int out_size, void* d_ws, size_t ws_size,
                              hipStream_t stream) {
    (void)in_sizes; (void)n_in; (void)out_size; (void)ws_size;
    const float* x = (const float*)d_in[0];
    const float* W = (const float*)d_in[1];
    float* out = (float*)d_out;

    char* ws = (char*)d_ws;
    _Float16* xh   = (_Float16*)(ws + XH_OFF);
    _Float16* wnt  = (_Float16*)(ws + WT_OFF);
    float*    xn   = (float*)(ws + XN_OFF);
    float*    qsum = (float*)(ws + QS_OFF);

    prep_x_kernel<<<BATCH / 8, 256, 0, stream>>>(x, xh, xn, qsum);
    prep_w_kernel<<<NPAD / 8, 256, 0, stream>>>(W, wnt);

    dim3 grid(NPAD / 128, BATCH / 128);   // 79 x 64
    gemm_kernel<1><<<grid, 256, 0, stream>>>(xh, wnt, xn, qsum, out);
    gemm_kernel<2><<<grid, 256, 0, stream>>>(xh, wnt, xn, qsum, out);
}
